// PairwiseMLPLinkPredictor_43490838839926
// MI455X (gfx1250) — matrix operand order repaired and checked by reading, not compiled, not run on hardware
//
#include <hip/hip_runtime.h>
#include <stdint.h>

// ---------------- problem constants ----------------
#define E_PAIRS   500000
#define IN_CH     256
#define H1_CH     256
#define H2_CH     128
#define M_TILES   (E_PAIRS / 16)       // 31250, E divides exactly by 16

// ---------------- WMMA types ----------------
typedef __attribute__((ext_vector_type(16))) __bf16 v16bf;
typedef __attribute__((ext_vector_type(8)))  float  v8f;

union ABHalves { uint4 q[2]; v16bf v; };

// ---------------- dynamic LDS layout (bytes) ----------------
// W1 swizzled bf16 tiles : 8 kb-tiles x 16 n-tiles x (32 lanes * 16 halves * 2B)
#define OFF_W1   0            // 131072 B
#define OFF_W2   131072       //  65536 B  (8 x 8 tiles)
#define OFF_B1   196608       //   1024 B  (256 f32)
#define OFF_B2   197632       //    512 B  (128 f32)
#define OFF_W3   198144       //    512 B  (128 f32)
#define OFF_B3   198656       //    256 B  (1 f32 + pad)
#define OFF_A    198912       // 4 waves x 8192 B  (16x256 bf16 feats; reused for layer-3 partials)
#define OFF_O    231680       // 4 waves x 8192 B  (16x256 bf16 layer-1 out)
#define SMEM_TOTAL 264448     // ~258 KB < 320 KB WGP LDS

__device__ __forceinline__ unsigned short f2bf(float f) {
    union { float f; unsigned u; } c; c.f = f;
    unsigned r = c.u + 0x7FFFu + ((c.u >> 16) & 1u);   // RNE
    return (unsigned short)(r >> 16);
}

extern "C" __global__ __launch_bounds__(128, 1)
void pairmlp_wmma(const float* __restrict__ x,
                  const long long* __restrict__ pairs,
                  const float* __restrict__ W1, const float* __restrict__ b1,
                  const float* __restrict__ W2, const float* __restrict__ b2,
                  const float* __restrict__ W3, const float* __restrict__ b3,
                  float* __restrict__ out)
{
    extern __shared__ unsigned char smem[];
    unsigned short* sW1 = (unsigned short*)(smem + OFF_W1);
    unsigned short* sW2 = (unsigned short*)(smem + OFF_W2);
    float* sB1 = (float*)(smem + OFF_B1);
    float* sB2 = (float*)(smem + OFF_B2);
    float* sW3 = (float*)(smem + OFF_W3);
    float* sB3 = (float*)(smem + OFF_B3);

    const int tid = threadIdx.x;

    // ---------- preload weights into LDS, pre-swizzled into WMMA B-operand layout ----------
    // B tile (32K x 16N, bf16): lane = (hi<<4)|n  (hi = k32>>4), element e = k32&15,
    // stored so each lane's 16 halves are contiguous: off = tile*512 + lane*16 + e.
    for (int idx4 = tid; idx4 < (IN_CH * H1_CH) / 4; idx4 += blockDim.x) {
        const int k  = idx4 >> 6;              // 64 float4 per 256-wide row
        const int n0 = (idx4 & 63) << 2;
        const float4 w = ((const float4*)W1)[idx4];
        const int kb = k >> 5, kr = k & 31;
        const int hh = kr >> 4, e = kr & 15;
        const float wv[4] = { w.x, w.y, w.z, w.w };
        #pragma unroll
        for (int c = 0; c < 4; ++c) {
            const int n = n0 + c;
            sW1[(kb * 16 + (n >> 4)) * 512 + (((hh << 4) | (n & 15)) << 4) + e] = f2bf(wv[c]);
        }
    }
    for (int idx4 = tid; idx4 < (H1_CH * H2_CH) / 4; idx4 += blockDim.x) {
        const int k  = idx4 >> 5;              // 32 float4 per 128-wide row
        const int n0 = (idx4 & 31) << 2;
        const float4 w = ((const float4*)W2)[idx4];
        const int kb = k >> 5, kr = k & 31;
        const int hh = kr >> 4, e = kr & 15;
        const float wv[4] = { w.x, w.y, w.z, w.w };
        #pragma unroll
        for (int c = 0; c < 4; ++c) {
            const int n = n0 + c;
            sW2[(kb * 8 + (n >> 4)) * 512 + (((hh << 4) | (n & 15)) << 4) + e] = f2bf(wv[c]);
        }
    }
    for (int i = tid; i < H1_CH; i += blockDim.x) sB1[i] = b1[i];
    for (int i = tid; i < H2_CH; i += blockDim.x) { sB2[i] = b2[i]; sW3[i] = W3[i]; }
    if (tid == 0) sB3[0] = b3[0];
    __syncthreads();

    const int waveId = tid >> 5;               // wave32
    const int lane   = tid & 31;
    const int m      = lane & 15;              // row (A) or column (B/C) within 16
    const int hi     = lane >> 4;

    unsigned short* sA = (unsigned short*)(smem + OFF_A) + waveId * 4096; // 16x256 bf16
    unsigned short* sO = (unsigned short*)(smem + OFF_O) + waveId * 4096; // 16x256 bf16
    float*          sP = (float*)(smem + OFF_A) + waveId * 2048;          // 16x16 f32 (reuse A)

    const float bias3 = sB3[0];
    const int stride = gridDim.x << 2;         // 4 waves per block

    for (int tile = blockIdx.x * 4 + waveId; tile < M_TILES; tile += stride) {
        asm volatile("" ::: "memory");
        const int row0 = tile << 4;

        // ---------- gather + elementwise product -> bf16 A staging (row-major 16x256) ----------
        // Each lane owns 8 columns [lane*8, lane*8+8).
        #pragma unroll 4
        for (int r = 0; r < 16; ++r) {
            const int p = row0 + r;
            const long long i = pairs[2 * p];
            const long long j = pairs[2 * p + 1];
            if (r + 2 < 16) {   // prefetch the gather 2 rows ahead (global_prefetch_b8)
                const long long i2 = pairs[2 * (p + 2)];
                const long long j2 = pairs[2 * (p + 2) + 1];
                __builtin_prefetch(x + (size_t)i2 * IN_CH + lane * 8, 0, 3);
                __builtin_prefetch(x + (size_t)j2 * IN_CH + lane * 8, 0, 3);
            }
            const float4* xi = (const float4*)(x + (size_t)i * IN_CH) + lane * 2;
            const float4* xj = (const float4*)(x + (size_t)j * IN_CH) + lane * 2;
            const float4 a0 = xi[0], a1 = xi[1];
            const float4 c0 = xj[0], c1 = xj[1];
            union { uint4 q; unsigned short h[8]; } u;
            u.h[0] = f2bf(a0.x * c0.x); u.h[1] = f2bf(a0.y * c0.y);
            u.h[2] = f2bf(a0.z * c0.z); u.h[3] = f2bf(a0.w * c0.w);
            u.h[4] = f2bf(a1.x * c1.x); u.h[5] = f2bf(a1.y * c1.y);
            u.h[6] = f2bf(a1.z * c1.z); u.h[7] = f2bf(a1.w * c1.w);
            *(uint4*)(sA + r * IN_CH + lane * 8) = u.q;
        }
        asm volatile("s_wait_dscnt 0x0" ::: "memory");

        // ---------- hoist entire A panel (16x256 bf16 = 8 WMMA tiles = 64 VGPRs) ----------
        // A layout (bf16 16x32): lane's halves = row m, K runs [kb*32+hi*16,+16).
        ABHalves aT[8];
        #pragma unroll
        for (int kb = 0; kb < 8; ++kb) {
            const unsigned short* ap = sA + m * IN_CH + kb * 32 + hi * 16;
            aT[kb].q[0] = *(const uint4*)(ap);
            aT[kb].q[1] = *(const uint4*)(ap + 8);
        }

        // ---------- layer 1: [16x256] x [256x256] + b1, ReLU -> sO ----------
        // 4 N-tiles per pass -> 4 independent WMMA accumulation chains (no TLP at 1 wave/SIMD).
        #pragma unroll 1
        for (int nt0 = 0; nt0 < 16; nt0 += 4) {
            v8f acc[4];
            #pragma unroll
            for (int t = 0; t < 4; ++t) {
                const float bv = sB1[(nt0 + t) * 16 + m];
                acc[t] = (v8f){ bv, bv, bv, bv, bv, bv, bv, bv };
            }
            #pragma unroll
            for (int kb = 0; kb < 8; ++kb) {
                #pragma unroll
                for (int t = 0; t < 4; ++t) {
                    ABHalves ub;
                    const unsigned short* bp = sW1 + (kb * 16 + nt0 + t) * 512 + lane * 16;
                    ub.q[0] = *(const uint4*)(bp);
                    ub.q[1] = *(const uint4*)(bp + 8);
                    acc[t] = __builtin_amdgcn_wmma_f32_16x16x32_bf16(
                                 false, aT[kb].v, false, ub.v, (short)0, acc[t], false, false);
                }
            }
            #pragma unroll
            for (int t = 0; t < 4; ++t) {
                #pragma unroll
                for (int v = 0; v < 8; ++v) {  // C layout: VGPR v -> row v (lanes<16) / v+8
                    const float f = fmaxf(acc[t][v], 0.0f);
                    sO[(v + (hi << 3)) * H1_CH + (nt0 + t) * 16 + m] = f2bf(f);
                }
            }
        }
        asm volatile("s_wait_dscnt 0x0" ::: "memory");

        // ---------- layer 2: [16x256] x [256x128] + b2 ; A panel + all 8 N-tiles in registers ----------
        ABHalves a2[8];
        #pragma unroll
        for (int kb = 0; kb < 8; ++kb) {
            const unsigned short* ap = sO + m * H1_CH + kb * 32 + hi * 16;
            a2[kb].q[0] = *(const uint4*)(ap);
            a2[kb].q[1] = *(const uint4*)(ap + 8);
        }
        v8f c2[8];
        #pragma unroll
        for (int nt = 0; nt < 8; ++nt) {
            const float bv = sB2[nt * 16 + m];
            c2[nt] = (v8f){ bv, bv, bv, bv, bv, bv, bv, bv };
        }
        #pragma unroll 1
        for (int kb = 0; kb < 8; ++kb) {
            #pragma unroll
            for (int nt = 0; nt < 8; ++nt) {   // 8 independent chains, A tile reused 8x
                ABHalves ub;
                const unsigned short* bp = sW2 + (kb * 8 + nt) * 512 + lane * 16;
                ub.q[0] = *(const uint4*)(bp);
                ub.q[1] = *(const uint4*)(bp + 8);
                c2[nt] = __builtin_amdgcn_wmma_f32_16x16x32_bf16(
                             false, a2[kb].v, false, ub.v, (short)0, c2[nt], false, false);
            }
        }

        // ---------- layer 3: ReLU(h2) . W3 + b3  (VALU dot + 16-lane LDS reduction) ----------
        float psum[8] = { 0, 0, 0, 0, 0, 0, 0, 0 };
        #pragma unroll
        for (int nt = 0; nt < 8; ++nt) {
            const float w = sW3[nt * 16 + m];
            #pragma unroll
            for (int v = 0; v < 8; ++v)
                psum[v] += fmaxf(c2[nt][v], 0.0f) * w;
        }
        #pragma unroll
        for (int v = 0; v < 8; ++v)
            sP[(v + (hi << 3)) * 16 + m] = psum[v];
        asm volatile("s_wait_dscnt 0x0" ::: "memory");

        if (lane < 16) {
            const float4* rp = (const float4*)(sP + lane * 16);
            const float4 s0 = rp[0], s1 = rp[1], s2 = rp[2], s3 = rp[3];
            const float s = (s0.x + s0.y + s0.z + s0.w) + (s1.x + s1.y + s1.z + s1.w)
                          + (s2.x + s2.y + s2.z + s2.w) + (s3.x + s3.y + s3.z + s3.w);
            out[row0 + lane] = s + bias3;
        }
    }
}

extern "C" void kernel_launch(void* const* d_in, const int* in_sizes, int n_in,
                              void* d_out, int out_size, void* d_ws, size_t ws_size,
                              hipStream_t stream) {
    const float*      x     = (const float*)d_in[0];
    // d_in[1] = edge_index (unused by the reference computation)
    const long long*  pairs = (const long long*)d_in[2];
    const float* W1 = (const float*)d_in[3];
    const float* b1 = (const float*)d_in[4];
    const float* W2 = (const float*)d_in[5];
    const float* b2 = (const float*)d_in[6];
    const float* W3 = (const float*)d_in[7];
    const float* b3 = (const float*)d_in[8];
    float* out = (float*)d_out;

    (void)in_sizes; (void)n_in; (void)out_size; (void)d_ws; (void)ws_size;

    // ~258 KB dynamic LDS per workgroup (WGP has 320 KB)
    (void)hipFuncSetAttribute((const void*)pairmlp_wmma,
                              hipFuncAttributeMaxDynamicSharedMemorySize, SMEM_TOTAL);

    dim3 grid(1024), block(128);   // 4 wave32 per block, persistent grid-stride over 31250 tiles
    pairmlp_wmma<<<grid, block, SMEM_TOTAL, stream>>>(x, pairs, W1, b1, W2, b2, W3, b3, out);
}
